// DeTr_32031866094241
// MI455X (gfx1250) — matrix operand order repaired and checked by reading, not compiled, not run on hardware
//
#include <hip/hip_runtime.h>
#include <hip/hip_bf16.h>
#include <math.h>

#define B_   2
#define CIN_ 1024
#define H_   60
#define W_   60
#define D_   512
#define NH_  8
#define NP_  9
#define DH_  64
#define N_   3600            // H_*W_
#define QT_  (N_ / 16)       // 225 q-tiles
#define INV_TEMP 10.0f
#define ATT_WT_  0.5f

typedef __attribute__((ext_vector_type(16))) _Float16 v16h;
typedef __attribute__((ext_vector_type(8)))  float    v8f;

// ---------------- WMMA fragment loaders (contiguous k order k=16*(l>>4)+j, same for A and B) ----------------
// A-matrix 16x32 f16: lane l holds row M=row0+(l&15); koff=16*(l>>4);
// element j -> K=koff+j.
__device__ __forceinline__ v16h load_a_f32(const float* A, int lda, int row0, int k0, int lane) {
  const float* p = A + (size_t)(row0 + (lane & 15)) * lda + k0 + ((lane >> 4) << 4);
  v16h f;
#pragma unroll
  for (int i = 0; i < 8; ++i) f[i] = (_Float16)p[i];
#pragma unroll
  for (int i = 0; i < 8; ++i) f[8 + i] = (_Float16)p[8 + i];
  return f;
}
__device__ __forceinline__ v16h load_a_h(const _Float16* A, int lda, int row0, int k0, int lane) {
  const _Float16* p = A + (size_t)(row0 + (lane & 15)) * lda + k0 + ((lane >> 4) << 4);
  v16h f;
#pragma unroll
  for (int i = 0; i < 8; ++i) f[i] = p[i];
#pragma unroll
  for (int i = 0; i < 8; ++i) f[8 + i] = p[8 + i];
  return f;
}
// B-matrix 32x16 f16: lane l holds col N=col0+(l&15); kbase=16*(l>>4);
// element j -> K=kbase+j.  Standard row-major B[k][n] source (strided), N clamped.
__device__ __forceinline__ v16h load_b_f32(const float* Bm, int ldb, int k0, int col0, int Nmax, int lane) {
  int nn = col0 + (lane & 15); if (nn > Nmax - 1) nn = Nmax - 1;
  const float* p = Bm + (size_t)(k0 + ((lane >> 4) << 4)) * ldb + nn;
  v16h f;
#pragma unroll
  for (int j = 0; j < 16; ++j) f[j] = (_Float16)p[(size_t)j * ldb];
  return f;
}
// B-matrix fragment from "transposed" source Bt[n][k] (contiguous in k)
__device__ __forceinline__ v16h load_bt_h(const _Float16* Bt, int ldt, int col0, int k0, int lane) {
  const _Float16* p = Bt + (size_t)(col0 + (lane & 15)) * ldt + k0 + ((lane >> 4) << 4);
  v16h f;
#pragma unroll
  for (int j = 0; j < 16; ++j) f[j] = p[j];
  return f;
}
__device__ __forceinline__ v16h load_bt_f32c(const float* Bt, int ldt, int col0, int k0, int kmax, int lane) {
  const float* p = Bt + (size_t)(col0 + (lane & 15)) * ldt;
  int kb = k0 + ((lane >> 4) << 4);
  v16h f;
#pragma unroll
  for (int j = 0; j < 16; ++j) { int k = kb + j; if (k > kmax - 1) k = kmax - 1; f[j] = (_Float16)p[k]; }
  return f;
}

__device__ __forceinline__ float wave_reduce_sum(float v) {
#pragma unroll
  for (int off = 16; off > 0; off >>= 1) v += __shfl_xor(v, off, 32);
  return v;
}

// ---------------- Generic register-tiled GEMM: C[M,N] = relu?(A[M,K]*B[K,N] + bias) ----------------
// Each wave computes a 32x32 output tile (4 WMMAs per 32-K chunk, 4 fragment loads).
// M%32==0, K%32==0, N tail-guarded (clamped loads, guarded stores).
__global__ void detr_gemm_kernel(const float* __restrict__ A, const float* __restrict__ Bm,
                                 const float* __restrict__ bias, float* __restrict__ C,
                                 int M, int N, int K, int lda, int ldb, int ldc, int relu)
{
  int lane = threadIdx.x & 31;
  int wave = blockIdx.x * (blockDim.x >> 5) + (threadIdx.x >> 5);
  int ntiles = (N + 31) >> 5;
  int total  = (M >> 5) * ntiles;
  if (wave >= total) return;                 // wave-uniform: EXEC stays all-1s for WMMA
  int mt = wave / ntiles, nt = wave - mt * ntiles;
  int row0 = mt << 5, col0 = nt << 5;
  v8f acc00 = {}, acc01 = {}, acc10 = {}, acc11 = {};
  for (int k0 = 0; k0 < K; k0 += 32) {
    if (k0 + 32 < K) {   // prefetch next K-chunk lines (global_prefetch_b8)
      __builtin_prefetch(A + (size_t)(row0 + (lane & 15)) * lda + k0 + 32 + ((lane >> 4) << 3), 0, 3);
      __builtin_prefetch(Bm + (size_t)(k0 + 32 + ((lane >> 4) << 4)) * ldb + col0 + (lane & 15), 0, 3);
    }
    v16h a0 = load_a_f32(A, lda, row0,      k0, lane);
    v16h a1 = load_a_f32(A, lda, row0 + 16, k0, lane);
    v16h b0 = load_b_f32(Bm, ldb, k0, col0,      N, lane);
    v16h b1 = load_b_f32(Bm, ldb, k0, col0 + 16, N, lane);
    acc00 = __builtin_amdgcn_wmma_f32_16x16x32_f16(false, a0, false, b0, (short)0, acc00, false, false);
    acc01 = __builtin_amdgcn_wmma_f32_16x16x32_f16(false, a0, false, b1, (short)0, acc01, false, false);
    acc10 = __builtin_amdgcn_wmma_f32_16x16x32_f16(false, a1, false, b0, (short)0, acc10, false, false);
    acc11 = __builtin_amdgcn_wmma_f32_16x16x32_f16(false, a1, false, b1, (short)0, acc11, false, false);
  }
  int nn0 = col0 + (lane & 15);
  int nn1 = nn0 + 16;
  int mb  = (lane >> 4) << 3;
  if (nn0 < N) {
    float bv = bias ? bias[nn0] : 0.f;
#pragma unroll
    for (int i = 0; i < 8; ++i) {
      float v0 = acc00[i] + bv; if (relu) v0 = fmaxf(v0, 0.f);
      C[(size_t)(row0 + mb + i) * ldc + nn0] = v0;
      float v1 = acc10[i] + bv; if (relu) v1 = fmaxf(v1, 0.f);
      C[(size_t)(row0 + 16 + mb + i) * ldc + nn0] = v1;
    }
  }
  if (nn1 < N) {
    float bv = bias ? bias[nn1] : 0.f;
#pragma unroll
    for (int i = 0; i < 8; ++i) {
      float v0 = acc01[i] + bv; if (relu) v0 = fmaxf(v0, 0.f);
      C[(size_t)(row0 + mb + i) * ldc + nn1] = v0;
      float v1 = acc11[i] + bv; if (relu) v1 = fmaxf(v1, 0.f);
      C[(size_t)(row0 + 16 + mb + i) * ldc + nn1] = v1;
    }
  }
}

// ---------------- l2-normalize fea[b][d][p] over d, store transposed f16 [b][p][d] ----------------
__global__ void detr_norm_t_kernel(const float* __restrict__ fea, _Float16* __restrict__ outT)
{
  int lane = threadIdx.x & 31;
  int wv = blockIdx.x * (blockDim.x >> 5) + (threadIdx.x >> 5);
  if (wv >= B_ * N_) return;
  int b = wv / N_, p = wv - b * N_;
  const float* col = fea + (size_t)b * D_ * N_ + p;
  float ss = 0.f;
  for (int d = lane; d < D_; d += 32) { float x = col[(size_t)d * N_]; ss += x * x; }
  ss = wave_reduce_sum(ss);
  float inv = 1.f / fmaxf(sqrtf(ss), 1e-12f);
  _Float16* orow = outT + (size_t)wv * D_;
  for (int d = lane; d < D_; d += 32) orow[d] = (_Float16)(col[(size_t)d * N_] * inv);
}

// ---------------- sine positional embedding ----------------
__device__ __forceinline__ float detr_pe(int p, int d) {
  int row = p / W_, col = p - row * W_;
  float v; int dd = d;
  if (d < 256) { v = (row + 1) * (2.0f * 3.14159265358979f) / (H_ + 1e-6f); }
  else         { v = (col + 1) * (2.0f * 3.14159265358979f) / (W_ + 1e-6f); dd = d - 256; }
  int j = dd >> 1;
  float freq = __powf(10000.f, (float)(2 * j) / 256.f);
  float ang = v / freq;
  return (dd & 1) ? __cosf(ang) : __sinf(ang);
}

// q[b][p][d] = fq_fea[b][d][p] + pe(p,d)
__global__ void detr_build_q_kernel(const float* __restrict__ fq_fea, float* __restrict__ q)
{
  int t = blockIdx.x * blockDim.x + threadIdx.x;
  if (t >= B_ * N_ * D_) return;
  int d = t & (D_ - 1);
  int r = t >> 9;               // D_ = 512
  int p = r % N_, b = r / N_;
  q[t] = fq_fea[(size_t)b * D_ * N_ + (size_t)d * N_ + p] + detr_pe(p, d);
}

// ---------------- flash cross-correlation attention ----------------
// ca[b][q][d] = sum_s softmax_s(fqn_q . fsn_s / TEMP) * f_s[b][d][s]
__global__ void detr_flash_corr_kernel(const _Float16* __restrict__ fqn, const _Float16* __restrict__ fsn,
                                       const float* __restrict__ f_s, float* __restrict__ ca)
{
  __shared__ float S[16][128];
  __shared__ float mrow[16], lrow[16], frow[16];
  int lane = threadIdx.x & 31;
  int w = threadIdx.x >> 5;
  int b = blockIdx.x / QT_;
  int qt = blockIdx.x - b * QT_;
  int q0 = qt << 4;
  const _Float16* Q  = fqn + (size_t)b * N_ * D_;   // [p][d]
  const _Float16* Kt = fsn + (size_t)b * N_ * D_;   // [s][d]
  const float*    V  = f_s + (size_t)b * D_ * N_;   // [d][s] -> Bt[N=d][k=s], ldt=N_
  if (threadIdx.x < 16) { mrow[threadIdx.x] = -1e30f; lrow[threadIdx.x] = 0.f; }
  v8f accO[4];
#pragma unroll
  for (int j = 0; j < 4; ++j) accO[j] = (v8f){};
  __syncthreads();

  const int SCH = 128;
  const int nchunks = (N_ + SCH - 1) / SCH;         // 29 (last chunk: only wave 0 valid)
  for (int ch = 0; ch < nchunks; ++ch) {
    int s0 = ch * SCH;
    int sw = s0 + (w << 4);
    v8f accS = {};
    if (sw < N_) {                                  // wave-uniform guard
      for (int k0 = 0; k0 < D_; k0 += 32) {
        if (k0 + 32 < D_) {
          __builtin_prefetch(Q  + (size_t)(q0 + (lane & 15)) * D_ + k0 + 32, 0, 3);
          __builtin_prefetch(Kt + (size_t)(sw + (lane & 15)) * D_ + k0 + 32, 0, 3);
        }
        v16h a  = load_a_h (Q,  D_, q0, k0, lane);
        v16h bb = load_bt_h(Kt, D_, sw, k0, lane);
        accS = __builtin_amdgcn_wmma_f32_16x16x32_f16(false, a, false, bb, (short)0, accS, false, false);
      }
    }
    {   // stage scaled scores into LDS, -inf padding past n
      int nn = lane & 15;
      int mb = (lane >> 4) << 3;
      int scol = (w << 4) + nn;
      bool ok = (sw < N_) && ((sw + nn) < N_);
#pragma unroll
      for (int i = 0; i < 8; ++i)
        S[mb + i][scol] = ok ? accS[i] * INV_TEMP : -1e30f;
    }
    __syncthreads();
    if (threadIdx.x < 16) {                         // online softmax stats per row
      int r = threadIdx.x;
      float mx = mrow[r];
      for (int c = 0; c < SCH; ++c) mx = fmaxf(mx, S[r][c]);
      float f = __expf(mrow[r] - mx);
      float l = lrow[r] * f;
      for (int c = 0; c < SCH; ++c) { float pr = __expf(S[r][c] - mx); S[r][c] = pr; l += pr; }
      mrow[r] = mx; lrow[r] = l; frow[r] = f;
    }
    __syncthreads();
    {   // rescale running accumulator
      int mb = (lane >> 4) << 3;
#pragma unroll
      for (int j = 0; j < 4; ++j)
#pragma unroll
        for (int i = 0; i < 8; ++i) accO[j][i] *= frow[mb + i];
    }
    int dbase = w << 6;                              // wave owns 64 columns of D
    for (int kc = 0; kc < SCH; kc += 32) {
      v16h pa = load_a_f32(&S[0][0], SCH, 0, kc, lane);
#pragma unroll
      for (int j = 0; j < 4; ++j) {
        v16h vb = load_bt_f32c(V, N_, dbase + (j << 4), s0 + kc, N_, lane);
        accO[j] = __builtin_amdgcn_wmma_f32_16x16x32_f16(false, pa, false, vb, (short)0, accO[j], false, false);
      }
    }
    __syncthreads();
  }
  {   // normalize by softmax denominator and store ca[b][q][d]
    int mb = (lane >> 4) << 3;
    int nn = lane & 15;
    float* Crow = ca + ((size_t)b * N_ + q0) * D_;
#pragma unroll
    for (int j = 0; j < 4; ++j)
#pragma unroll
      for (int i = 0; i < 8; ++i) {
        float denom = fmaxf(lrow[mb + i], 1e-30f);
        Crow[(size_t)(mb + i) * D_ + ((w << 6) + (j << 4) + nn)] = accO[j][i] / denom;
      }
  }
}

// ---------------- f_q1 = l2n(f_q) + l2n(ca)*ATT_WT ; also emit ca transposed to d_out ----------------
__global__ void detr_fq1_kernel(const float* __restrict__ f_q, const float* __restrict__ ca,
                                float* __restrict__ f_q1, float* __restrict__ out_ca)
{
  int lane = threadIdx.x & 31;
  int wv = blockIdx.x * (blockDim.x >> 5) + (threadIdx.x >> 5);
  if (wv >= B_ * N_) return;
  int b = wv / N_, p = wv - b * N_;
  const float* qc = f_q + (size_t)b * D_ * N_ + p;   // stride N_
  const float* cr = ca + (size_t)wv * D_;            // contiguous
  float ssq = 0.f, ssc = 0.f;
  for (int d = lane; d < D_; d += 32) {
    float a = qc[(size_t)d * N_]; ssq += a * a;
    float c = cr[d];              ssc += c * c;
  }
  ssq = wave_reduce_sum(ssq);
  ssc = wave_reduce_sum(ssc);
  float iq = 1.f / fmaxf(sqrtf(ssq), 1e-12f);
  float ic = 1.f / fmaxf(sqrtf(ssc), 1e-12f);
  float* orow = f_q1 + (size_t)wv * D_;
  float* ocol = out_ca + (size_t)b * D_ * N_ + p;
  for (int d = lane; d < D_; d += 32) {
    float c = cr[d];
    orow[d] = qc[(size_t)d * N_] * iq + ATT_WT_ * c * ic;
    ocol[(size_t)d * N_] = c;
  }
}

// ---------------- attention-weight softmax over NP=9 per (pos, head) ----------------
__global__ void detr_aw_softmax_kernel(float* __restrict__ awb)
{
  int t = blockIdx.x * blockDim.x + threadIdx.x;
  if (t >= B_ * N_ * NH_) return;
  float* g = awb + (size_t)t * NP_;
  float mx = g[0];
#pragma unroll
  for (int i = 1; i < NP_; ++i) mx = fmaxf(mx, g[i]);
  float s = 0.f; float e[NP_];
#pragma unroll
  for (int i = 0; i < NP_; ++i) { e[i] = __expf(g[i] - mx); s += e[i]; }
  float inv = 1.f / s;
#pragma unroll
  for (int i = 0; i < NP_; ++i) g[i] = e[i] * inv;
}

// ---------------- deformable bilinear sampling + weighted sum ----------------
__device__ __forceinline__ float detr_corner(const float* vbase, int ix, int iy, float wt) {
  if (ix < 0 || ix >= W_ || iy < 0 || iy >= H_) return 0.f;
  return vbase[(size_t)(iy * W_ + ix) * D_] * wt;
}
__global__ void detr_sample_kernel(const float* __restrict__ value, const float* __restrict__ offsets,
                                   const float* __restrict__ aw, float* __restrict__ attnout)
{
  int t = blockIdx.x * blockDim.x + threadIdx.x;
  if (t >= B_ * N_ * NH_ * DH_) return;
  int dh = t & (DH_ - 1);
  int r = t >> 6;
  int h = r & (NH_ - 1);
  r >>= 3;
  int p = r % N_, b = r / N_;
  int row = p / W_, col = p - row * W_;
  float refx = (col + 0.5f) / W_;
  float refy = (row + 0.5f) / H_;
  const float* off = offsets + ((size_t)(b * N_ + p)) * (NH_ * NP_ * 2) + h * NP_ * 2;
  const float* a   = aw      + ((size_t)(b * N_ + p)) * (NH_ * NP_)     + h * NP_;
  const float* vbase = value + (size_t)b * N_ * D_ + h * DH_ + dh;   // indexed by pos*D_
  float acc = 0.f;
#pragma unroll
  for (int pt = 0; pt < NP_; ++pt) {
    float x = (refx + off[pt * 2 + 0] / W_) * W_ - 0.5f;
    float y = (refy + off[pt * 2 + 1] / H_) * H_ - 0.5f;
    float x0 = floorf(x), y0 = floorf(y);
    float wx1 = x - x0, wx0 = 1.f - wx1;
    float wy1 = y - y0, wy0 = 1.f - wy1;
    int ix0 = (int)x0, iy0 = (int)y0;
    float s = detr_corner(vbase, ix0,     iy0,     wx0 * wy0)
            + detr_corner(vbase, ix0 + 1, iy0,     wx1 * wy0)
            + detr_corner(vbase, ix0,     iy0 + 1, wx0 * wy1)
            + detr_corner(vbase, ix0 + 1, iy0 + 1, wx1 * wy1);
    acc += a[pt] * s;
  }
  attnout[((size_t)(b * N_ + p)) * D_ + h * DH_ + dh] = acc;
}

// ---------------- f_q2 = l2n(f_q1) + l2n(sa)*ATT_WT ; emit f_q2 and sa transposed ----------------
__global__ void detr_final_kernel(const float* __restrict__ f_q1, const float* __restrict__ sa,
                                  float* __restrict__ out_fq2, float* __restrict__ out_sa)
{
  int lane = threadIdx.x & 31;
  int wv = blockIdx.x * (blockDim.x >> 5) + (threadIdx.x >> 5);
  if (wv >= B_ * N_) return;
  int b = wv / N_, p = wv - b * N_;
  const float* r1 = f_q1 + (size_t)wv * D_;
  const float* rs = sa   + (size_t)wv * D_;
  float ss1 = 0.f, sss = 0.f;
  for (int d = lane; d < D_; d += 32) { float a = r1[d]; ss1 += a * a; float s = rs[d]; sss += s * s; }
  ss1 = wave_reduce_sum(ss1);
  sss = wave_reduce_sum(sss);
  float i1 = 1.f / fmaxf(sqrtf(ss1), 1e-12f);
  float is = 1.f / fmaxf(sqrtf(sss), 1e-12f);
  float* o1 = out_fq2 + (size_t)b * D_ * N_ + p;
  float* o2 = out_sa  + (size_t)b * D_ * N_ + p;
  for (int d = lane; d < D_; d += 32) {
    float s = rs[d];
    o1[(size_t)d * N_] = r1[d] * i1 + ATT_WT_ * s * is;
    o2[(size_t)d * N_] = s;
  }
}

// ---------------- host orchestration ----------------
extern "C" void kernel_launch(void* const* d_in, const int* in_sizes, int n_in,
                              void* d_out, int out_size, void* d_ws, size_t ws_size,
                              hipStream_t stream)
{
  (void)in_sizes; (void)n_in; (void)out_size; (void)ws_size;
  const float* fq_l3 = (const float*)d_in[0];
  const float* fs_l3 = (const float*)d_in[1];
  const float* f_q   = (const float*)d_in[2];
  const float* f_s   = (const float*)d_in[3];
  const float* conv_w= (const float*)d_in[4];
  const float* vp_w  = (const float*)d_in[5];
  const float* vp_b  = (const float*)d_in[6];
  const float* off_w = (const float*)d_in[7];
  const float* off_b = (const float*)d_in[8];
  const float* aw_w  = (const float*)d_in[9];
  const float* aw_b  = (const float*)d_in[10];
  const float* op_w  = (const float*)d_in[11];
  const float* op_b  = (const float*)d_in[12];

  const size_t BDN = (size_t)B_ * D_ * N_;   // 3,686,400
  float* out_fq2 = (float*)d_out;
  float* out_sa  = out_fq2 + BDN;
  float* out_ca  = out_sa + BDN;

  char* w = (char*)d_ws;
  auto take = [&](size_t bytes) { char* p = w; w += (bytes + 255) & ~(size_t)255; return p; };
  float*    fq_fea  = (float*)take(BDN * 4);
  float*    fs_fea  = (float*)take(BDN * 4);
  _Float16* fqn     = (_Float16*)take(BDN * 2);
  _Float16* fsn     = (_Float16*)take(BDN * 2);
  float*    qbuf    = (float*)take(BDN * 4);
  float*    f_q1    = (float*)take(BDN * 4);
  float*    sabuf   = (float*)take(BDN * 4);
  float*    offbuf  = (float*)take((size_t)B_ * N_ * NH_ * NP_ * 2 * 4);
  float*    awbuf   = (float*)take((size_t)B_ * N_ * NH_ * NP_ * 4);
  float*    cabuf   = fq_fea;   // fq_fea dead after q is built
  float*    attnout = fs_fea;   // fs_fea dead after normalization
  float*    value   = (float*)fqn; // fqn+fsn region (exact size) dead after flash

  auto gemm_blocks = [](int M, int N) { int waves = (M >> 5) * ((N + 31) >> 5); return (waves + 7) / 8; };

  // 1) 1x1 conv + ReLU: fea[b] = relu(conv_w[512x1024] * X[b][1024x3600])
  for (int b = 0; b < B_; ++b) {
    detr_gemm_kernel<<<gemm_blocks(D_, N_), 256, 0, stream>>>(
        conv_w, fq_l3 + (size_t)b * CIN_ * N_, nullptr, fq_fea + (size_t)b * D_ * N_,
        D_, N_, CIN_, CIN_, N_, N_, 1);
    detr_gemm_kernel<<<gemm_blocks(D_, N_), 256, 0, stream>>>(
        conv_w, fs_l3 + (size_t)b * CIN_ * N_, nullptr, fs_fea + (size_t)b * D_ * N_,
        D_, N_, CIN_, CIN_, N_, N_, 1);
  }

  // 2) l2-normalize + transpose to f16 [b][p][d]
  int nrmBlocks = (B_ * N_ + 7) / 8;
  detr_norm_t_kernel<<<nrmBlocks, 256, 0, stream>>>(fq_fea, fqn);
  detr_norm_t_kernel<<<nrmBlocks, 256, 0, stream>>>(fs_fea, fsn);

  // 3) q = fq_fea^T + sine positional embedding
  detr_build_q_kernel<<<(B_ * N_ * D_ + 255) / 256, 256, 0, stream>>>(fq_fea, qbuf);

  // 4) flash cross-correlation attention -> ca[b][p][d]
  detr_flash_corr_kernel<<<B_ * QT_, 256, 0, stream>>>(fqn, fsn, f_s, cabuf);

  // 5) f_q1 = l2n(f_q) + l2n(ca)*0.5 ; emit ca output
  detr_fq1_kernel<<<nrmBlocks, 256, 0, stream>>>(f_q, cabuf, f_q1, out_ca);

  // 6) value / offsets / attention-weight projections
  detr_gemm_kernel<<<gemm_blocks(B_ * N_, D_), 256, 0, stream>>>(
      f_q1, vp_w, vp_b, value, B_ * N_, D_, D_, D_, D_, D_, 0);
  detr_gemm_kernel<<<gemm_blocks(B_ * N_, NH_ * NP_ * 2), 256, 0, stream>>>(
      qbuf, off_w, off_b, offbuf, B_ * N_, NH_ * NP_ * 2, D_, D_, NH_ * NP_ * 2, NH_ * NP_ * 2, 0);
  detr_gemm_kernel<<<gemm_blocks(B_ * N_, NH_ * NP_), 256, 0, stream>>>(
      qbuf, aw_w, aw_b, awbuf, B_ * N_, NH_ * NP_, D_, D_, NH_ * NP_, NH_ * NP_, 0);
  detr_aw_softmax_kernel<<<(B_ * N_ * NH_ + 255) / 256, 256, 0, stream>>>(awbuf);

  // 7) deformable bilinear sampling + head mix
  detr_sample_kernel<<<(B_ * N_ * NH_ * DH_ + 255) / 256, 256, 0, stream>>>(value, offbuf, awbuf, attnout);

  // 8) output projection
  detr_gemm_kernel<<<gemm_blocks(B_ * N_, D_), 256, 0, stream>>>(
      attnout, op_w, op_b, sabuf, B_ * N_, D_, D_, D_, D_, D_, 0);

  // 9) f_q2 = l2n(f_q1) + l2n(sa)*0.5 ; emit f_q2 and sa outputs
  detr_final_kernel<<<nrmBlocks, 256, 0, stream>>>(f_q1, sabuf, out_fq2, out_sa);
}
